// SimpleAttention_48215302865416
// MI455X (gfx1250) — matrix operand order repaired and checked by reading, not compiled, not run on hardware
//
#include <hip/hip_runtime.h>

#define B_  8
#define SQ  2048
#define SK  2048
#define DK  128
#define DV  128

typedef __attribute__((ext_vector_type(16))) _Float16 v16h;
typedef __attribute__((ext_vector_type(8)))  float    v8f;
typedef __attribute__((ext_vector_type(4)))  _Float16 h4;
typedef __attribute__((ext_vector_type(4)))  float    f4;

// A-fragment (16x32 f16, row-major in memory): lane l (0..15) holds row m=l.
// half h==0: K = base+0..7 (VGPR0..3) and base+16..23 (VGPR4..7)
// half h==1: K = base+8..15           and base+24..31
__device__ __forceinline__ v16h load_a_frag(const float* rowp /* row start + kc */, int h) {
  const float* p = rowp + h * 8;
  v16h r;
#pragma unroll
  for (int i = 0; i < 8; i++) r[i] = (_Float16)p[i];
#pragma unroll
  for (int i = 0; i < 8; i++) r[8 + i] = (_Float16)p[16 + i];
  return r;
}

__global__ __launch_bounds__(256) void fused_attn_kernel(
    const float* __restrict__ Q, const float* __restrict__ K,
    const float* __restrict__ V, const int* __restrict__ mask,
    float* __restrict__ out, float* __restrict__ attn)
{
  // 16 query rows x 2048 keys of f16 scores/probs: 64KB of the 320KB WGP LDS.
  __shared__ _Float16 S[16 * SK];

  const int qtile = blockIdx.x;            // 0..SQ/16-1
  const int b     = blockIdx.y;            // 0..B-1
  const int wave  = threadIdx.x >> 5;      // 0..7
  const int lane  = threadIdx.x & 31;
  const int l     = lane & 15;
  const int h     = lane >> 4;
  const int qbase = qtile * 16;

  const float scale = 0.08838834764831845f; // 1/sqrt(128)

  // ---------------- Phase 1: S = mask(scale * Q K^T) -> LDS (f16) ----------------
  const float* Qb = Q + ((size_t)b * SQ + qbase) * DK;

  // Load the Q A-fragments once (4 chunks of K=32 cover DK=128); reused 16x.
  v16h aq[4];
#pragma unroll
  for (int kc = 0; kc < 4; kc++)
    aq[kc] = load_a_frag(Qb + l * DK + kc * 32, h);

  // Wave w owns k-tiles [w*16, w*16+16) -> LDS columns [256w, 256w+256).
  for (int kt = wave * 16; kt < wave * 16 + 16; kt++) {
    const float* Kb = K + ((size_t)b * SK + kt * 16) * DK;
    v8f c = {};
#pragma unroll
    for (int kc = 0; kc < 4; kc++) {
      // B-fragment: B[k][n] = K[n][k]; lane (l=n, half h) holds K=kc*32+h*8+0..7 and kc*32+16+h*8+0..7
      const float* kp = Kb + l * DK + kc * 32 + h * 8;
      v16h bm;
#pragma unroll
      for (int i = 0; i < 8; i++) bm[i] = (_Float16)kp[i];
#pragma unroll
      for (int i = 0; i < 8; i++) bm[8 + i] = (_Float16)kp[16 + i];
      c = __builtin_amdgcn_wmma_f32_16x16x32_f16(false, aq[kc], false, bm,
                                                 (short)0, c, false, false);
    }
    const int kbase = kt * 16;
#pragma unroll
    for (int r = 0; r < 8; r++) {
      const int m = r + h * 8;                  // C layout: VGPR r -> row r + 8*half
      float s = c[r] * scale;
      // Mask is single-use 134MB stream: non-temporal so it doesn't evict K/V in L2.
      int mk = __builtin_nontemporal_load(
                 mask + ((size_t)b * SQ + qbase + m) * SK + kbase + l);
      if (mk == 0) s = -30000.0f;               // finite in f16; exp underflows to 0
      S[m * SK + kbase + l] = (_Float16)s;
    }
  }
  __syncthreads();

  // ---------------- Phase 2: row softmax; write f32 weights to global ------------
  // Wave w handles rows 2w and 2w+1; each lane owns 4-element vectors (ds_load_b64).
#pragma unroll
  for (int rr = 0; rr < 2; rr++) {
    const int m = wave * 2 + rr;
    _Float16* row = S + m * SK;

    float mx = -1e30f;
    for (int i = lane * 4; i < SK; i += 128) {
      h4 v = *(const h4*)(row + i);
      mx = fmaxf(mx, fmaxf(fmaxf((float)v[0], (float)v[1]),
                           fmaxf((float)v[2], (float)v[3])));
    }
#pragma unroll
    for (int off = 16; off > 0; off >>= 1) mx = fmaxf(mx, __shfl_xor(mx, off, 32));

    float sum = 0.0f;
    for (int i = lane * 4; i < SK; i += 128) {
      h4 v = *(const h4*)(row + i);
      float e0 = __expf((float)v[0] - mx);
      float e1 = __expf((float)v[1] - mx);
      float e2 = __expf((float)v[2] - mx);
      float e3 = __expf((float)v[3] - mx);
      sum += (e0 + e1) + (e2 + e3);
      h4 e = { (_Float16)e0, (_Float16)e1, (_Float16)e2, (_Float16)e3 };
      *(h4*)(row + i) = e;
    }
#pragma unroll
    for (int off = 16; off > 0; off >>= 1) sum += __shfl_xor(sum, off, 32);
    const float inv = 1.0f / sum;

    float* grow = attn + ((size_t)b * SQ + qbase + m) * SK;
    for (int i = lane * 4; i < SK; i += 128) {
      h4 v = *(const h4*)(row + i);
      f4 w = { (float)v[0] * inv, (float)v[1] * inv,
               (float)v[2] * inv, (float)v[3] * inv };
      // Write-once 134MB stream: b128 non-temporal store, keep L2 for K/V.
      __builtin_nontemporal_store(w, (f4*)(grow + i));
      h4 p = { (_Float16)w[0], (_Float16)w[1], (_Float16)w[2], (_Float16)w[3] };
      *(h4*)(row + i) = p;    // normalized prob kept in LDS for PV
    }
  }
  __syncthreads();

  // ---------------- Phase 3: out = P @ V (WMMA, A from LDS) ----------------------
  {
    const int vbase = wave * 16;  // 8 waves x 16 = DV=128
    v8f c = {};
    for (int kc = 0; kc < SK; kc += 32) {
      // A-fragment from LDS (contiguous f16 -> 16B ds loads)
      const _Float16* ap = S + l * SK + kc + h * 8;
      v16h av;
#pragma unroll
      for (int i = 0; i < 8; i++) av[i] = ap[i];
#pragma unroll
      for (int i = 0; i < 8; i++) av[8 + i] = ap[16 + i];
      // B-fragment: B[k][n] = V[kc+k][vbase+n]; strided gather down V columns
      const float* vp = V + ((size_t)b * SK + kc + h * 8) * DV + vbase + l;
      v16h bv;
#pragma unroll
      for (int i = 0; i < 8; i++) bv[i] = (_Float16)vp[i * DV];
#pragma unroll
      for (int i = 0; i < 8; i++) bv[8 + i] = (_Float16)vp[(16 + i) * DV];
      c = __builtin_amdgcn_wmma_f32_16x16x32_f16(false, av, false, bv,
                                                 (short)0, c, false, false);
    }
#pragma unroll
    for (int r = 0; r < 8; r++) {
      const int m = r + h * 8;
      __builtin_nontemporal_store(
          c[r], out + ((size_t)b * SQ + qbase + m) * DV + vbase + l);
    }
  }
}

extern "C" void kernel_launch(void* const* d_in, const int* in_sizes, int n_in,
                              void* d_out, int out_size, void* d_ws, size_t ws_size,
                              hipStream_t stream) {
  const float* Q    = (const float*)d_in[0];
  const float* K    = (const float*)d_in[1];
  const float* V    = (const float*)d_in[2];
  const int*   mask = (const int*)d_in[3];

  float* out  = (float*)d_out;                       // [B,SQ,DV] first
  float* attn = out + (size_t)B_ * SQ * DV;          // then [B,SQ,SK]

  dim3 grid(SQ / 16, B_);
  fused_attn_kernel<<<grid, 256, 0, stream>>>(Q, K, V, mask, out, attn);
}
